// GRU_RGCN_62362925138251
// MI455X (gfx1250) — matrix operand order repaired and checked by reading, not compiled, not run on hardware
//
#include <hip/hip_runtime.h>
#include <hip/hip_fp16.h>
#include <stdint.h>

#define T_STEPS 64
#define NNODES  32
#define NREL    5
#define NEDGE   25
#define DIM     512
#define NGLOB   50000
#define NSENSE  25000
#define NTOT    75000          // NGLOB + NSENSE
#define ND      (NNODES*DIM)   // 16384

typedef __attribute__((ext_vector_type(16))) _Float16 v16h;
typedef __attribute__((ext_vector_type(8)))  float    v8f;

// monotone float <-> uint encoding so unsigned atomicMax gives float max
__device__ __forceinline__ unsigned enc_f32(float f) {
  unsigned u = __float_as_uint(f);
  return (u & 0x80000000u) ? ~u : (u | 0x80000000u);
}
__device__ __forceinline__ float dec_f32(unsigned e) {
  return (e & 0x80000000u) ? __uint_as_float(e ^ 0x80000000u)
                           : __uint_as_float(~e);
}

// ---------------------------------------------------------------- K0: init
// Convert the 6 static [512x512] weight matrices (5 conv_W + W_0) to f16 and
// zero the recurrent memory. Runs once per launch; deterministic.
__global__ void k0_init(const float* __restrict__ convW, const float* __restrict__ W0,
                        _Float16* __restrict__ Wh, float* __restrict__ mem) {
  int i = blockIdx.x * 256 + threadIdx.x;
  const int total = 6 * DIM * DIM;
  if (i < total) {
    float v = (i < NREL * DIM * DIM) ? convW[i] : W0[i - NREL * DIM * DIM];
    Wh[i] = (_Float16)v;
  }
  if (i < ND) mem[i] = 0.0f;
}

// ---------------------------------------------------------------- K1: gather
__global__ void k1_gather(const int* __restrict__ xidx, const float* __restrict__ X,
                          float* __restrict__ gx, _Float16* __restrict__ gxh, int t) {
  int i = blockIdx.x * 256 + threadIdx.x;       // i < 16384
  int n = i >> 9, d = i & 511;
  int row = xidx[t * NNODES + n];
  float v = X[(long)row * DIM + d];
  gx[i]  = v;
  gxh[i] = (_Float16)v;
}

// ---------------------------------------------------------------- K2: WMMA GEMMs
// xw[m] = gx(32x512) @ W_m(512x512) for m=0..5 (5 conv_W + W_0), f16 in / f32 out.
// Grid: 6 matrices x 8 column-tiles of 64.  Block: 256 thr = 8 waves; each wave
// owns one 16x16 D tile and loops K in steps of 32 via v_wmma_f32_16x16x32_f16.
// B K-slabs are staged with CDNA5 GLOBAL_LOAD_ASYNC_TO_LDS_B128 (ASYNCcnt).
#define SAS 520   // padded LDS row stride (halfs) to break bank conflicts
__global__ void k2_gemm(const _Float16* __restrict__ gxh, const _Float16* __restrict__ Wh,
                        float* __restrict__ xw) {
  __shared__ __align__(16) _Float16 sA[32 * SAS];   // whole A: 32x512
  __shared__ __align__(16) _Float16 sB[32 * 64];    // B K-slab: 32x64
  const int tid  = threadIdx.x;
  const int mat  = blockIdx.x >> 3;
  const int col0 = (blockIdx.x & 7) * 64;
  const _Float16* W = Wh + (size_t)mat * DIM * DIM;

  for (int idx = tid; idx < 2048; idx += 256) {       // stage A once
    int r = idx >> 6, j = idx & 63;
    *(uint4*)&sA[r * SAS + j * 8] = *(const uint4*)(gxh + r * DIM + j * 8);
  }

  const int wave = tid >> 5;
  const int lane = tid & 31;
  const int g    = lane >> 4;     // half-wave group
  const int ml   = lane & 15;
  const int tm   = wave & 1;      // tile row (rows 16*tm..16*tm+15)
  const int tn   = wave >> 1;     // tile col within the 64-wide slab
  v8f acc = {};

  const int br = tid >> 3, bj = tid & 7;              // this thread's B-stage slot
  const unsigned ldsoff = (unsigned)(uintptr_t)&sB[br * 64 + bj * 8];

  for (int k0 = 0; k0 < DIM; k0 += 32) {
    __syncthreads();                                  // prev-iter frag reads done
    {                                                 // async stage B slab 32x64
      unsigned long long gsrc =
          (unsigned long long)(uintptr_t)(W + (size_t)(k0 + br) * DIM + col0 + bj * 8);
      asm volatile("global_load_async_to_lds_b128 %0, %1, off"
                   :: "v"(ldsoff), "v"(gsrc) : "memory");
      asm volatile("s_wait_asynccnt 0x0" ::: "memory");
    }
    __syncthreads();
    v16h a, b;
    // A frag, contiguous K order: VGPR v holds K pair kb,kb+1 with kb = 16*g + 2*v
    #pragma unroll
    for (int v = 0; v < 8; ++v) {
      int kb = 16 * g + 2 * v;
      a[2 * v]     = sA[(16 * tm + ml) * SAS + k0 + kb];
      a[2 * v + 1] = sA[(16 * tm + ml) * SAS + k0 + kb + 1];
    }
    // B frag: lanes 0-15 cover K=0..15, lanes 16-31 cover K=16..31, col = ml
    #pragma unroll
    for (int j = 0; j < 16; ++j)
      b[j] = sB[(16 * g + j) * 64 + 16 * tn + ml];
    acc = __builtin_amdgcn_wmma_f32_16x16x32_f16(false, a, false, b,
                                                 (short)0, acc, false, false);
  }

  float* out = xw + (size_t)mat * ND;
  #pragma unroll
  for (int v = 0; v < 8; ++v) {        // D layout: row = v + 8*g, col = ml
    int row = 16 * tm + v + 8 * g;
    int col = col0 + 16 * tn + ml;
    out[row * DIM + col] = acc[v];
  }
}

// ---------------------------------------------------------------- K2.5: gate partials
// gate pre-activation = gx_flat(1x16384) @ gate_W(16384x512)
//                     + mem[0] (1x512)   @ gate_U(512x512)
// split over 64 blocks (block b: gate_W rows b*256.., gate_U rows b*8..).
// mem is read from the PREVIOUS step (final before this kernel runs), which
// moves the 1 MB gate_U matvec off the serial single-WGP K3 critical path.
__global__ void k25_gatep(const float* __restrict__ gx, const float* __restrict__ gateW,
                          const float* __restrict__ mem, const float* __restrict__ gateU,
                          float* __restrict__ gatep) {
  __shared__ float sx[256];
  int b = blockIdx.x, tid = threadIdx.x;
  sx[tid] = gx[b * 256 + tid];
  __syncthreads();
  int d0 = tid, d1 = tid + 256;
  float a0 = 0.f, a1 = 0.f;
  const float* Wp = gateW + (size_t)b * 256 * DIM;
  for (int i = 0; i < 256; ++i) {
    float xv = sx[i];
    const float* row = Wp + (size_t)i * DIM;
    a0 += xv * row[d0];
    a1 += xv * row[d1];
  }
  const float* Up = gateU + (size_t)b * 8 * DIM;
  #pragma unroll
  for (int i = 0; i < 8; ++i) {
    float mv = mem[b * 8 + i];                 // mem row 0 (prev step)
    const float* row = Up + (size_t)i * DIM;
    a0 += mv * row[d0];
    a1 += mv * row[d1];
  }
  gatep[b * DIM + d0] = a0;
  gatep[b * DIM + d1] = a1;
}

// ---------------------------------------------------------------- K3: graph + gate + GRU
// Single block of 512 threads: sym-normalized propagation, gate sigmoid,
// rgcn update (writes mem), h = relu(rgcn[0]), and per-step reset of red slots.
__global__ void k3_combine(const int* __restrict__ eidx, const float* __restrict__ xw,
                           const float* __restrict__ gatep,
                           float* __restrict__ mem, float* __restrict__ hbuf,
                           unsigned* __restrict__ red, int t) {
  __shared__ float sdinv[NREL * NNODES];
  __shared__ int   sedge[NREL * 2 * NEDGE];
  __shared__ float sgate[DIM];
  int tid = threadIdx.x;
  const int* ebase = eidx + (size_t)t * NREL * 2 * NEDGE;
  if (tid < NREL * 2 * NEDGE) sedge[tid] = ebase[tid];
  if (tid < NREL * NNODES) {                 // deg^{-1/2} incl. self loop
    int r = tid >> 5, n = tid & 31;
    const int* dstp = ebase + r * 2 * NEDGE + NEDGE;
    int cnt = 0;
    for (int e = 0; e < NEDGE; ++e) cnt += (dstp[e] == n);
    sdinv[tid] = rsqrtf(1.0f + (float)cnt);
  }
  if (tid == 0) {                            // per-step reset of reduction slots
    red[0] = 0u; red[1] = 0u;
    ((float*)red)[2] = 0.f; ((float*)red)[3] = 0.f;
  }
  __syncthreads();

  // conv + proposed: thread owns (node n, 32-wide column chunk)
  int n = tid >> 4, dbase = (tid & 15) << 5;
  float acc[32];
  #pragma unroll
  for (int j = 0; j < 32; ++j) acc[j] = 0.f;
  for (int r = 0; r < NREL; ++r) {
    float dn = sdinv[r * 32 + n];
    const float* xr = xw + (size_t)r * ND;
    {
      float s = dn * dn;                                  // self-loop term
      const float* p = xr + n * DIM + dbase;
      #pragma unroll
      for (int j = 0; j < 32; ++j) acc[j] += s * p[j];
    }
    const int* srcp = &sedge[r * 2 * NEDGE];
    const int* dstp = srcp + NEDGE;
    for (int e = 0; e < NEDGE; ++e) {
      if (dstp[e] == n) {
        int s = srcp[e];
        float norm = sdinv[r * 32 + s] * dn;
        const float* p = xr + s * DIM + dbase;
        #pragma unroll
        for (int j = 0; j < 32; ++j) acc[j] += norm * p[j];
      }
    }
  }
  {
    const float* p = xw + (size_t)5 * ND + n * DIM + dbase;  // + gx @ W_0
    #pragma unroll
    for (int j = 0; j < 32; ++j) acc[j] += p[j];
  }

  // gate[d] = sigmoid(sum of 64 partials), one d per thread
  {
    int d = tid;
    float gsum = 0.f;
    for (int b = 0; b < 64; ++b) gsum += gatep[b * DIM + d];
    sgate[d] = 1.0f / (1.0f + __expf(-gsum));
  }
  __syncthreads();

  // rgcn = g*proposed + (1-g)*mem ; mem <- rgcn ; h = relu(rgcn[0])
  #pragma unroll
  for (int j = 0; j < 32; ++j) {
    int d = dbase + j;
    float gd = sgate[d];
    int idx = n * DIM + d;
    float rv = gd * acc[j] + (1.0f - gd) * mem[idx];
    mem[idx] = rv;
    if (n == 0) hbuf[d] = fmaxf(rv, 0.0f);
  }
}

// ---------------------------------------------------------------- K4: classifier logits
// One wave32 per row, 8 rows per wave; b128 weight streaming + shuffle reduce.
__global__ void k4_logits(const float* __restrict__ hbuf,
                          const float* __restrict__ gw, const float* __restrict__ gb,
                          const float* __restrict__ sw, const float* __restrict__ sb,
                          float* __restrict__ z, unsigned* __restrict__ red) {
  __shared__ __align__(16) float sh[DIM];
  __shared__ unsigned smax[2];
  int tid = threadIdx.x;
  sh[tid] = hbuf[tid];
  sh[tid + 256] = hbuf[tid + 256];
  if (tid < 2) smax[tid] = 0u;
  __syncthreads();
  int wave = tid >> 5, lane = tid & 31;
  int rowbase = blockIdx.x * 64 + wave * 8;
  for (int rr = 0; rr < 8; ++rr) {
    int row = rowbase + rr;
    if (row >= NTOT) break;                   // wave-uniform
    const float* w; float bias; int slot;
    if (row < NGLOB) { w = gw + (size_t)row * DIM; bias = gb[row]; slot = 0; }
    else { int r2 = row - NGLOB; w = sw + (size_t)r2 * DIM; bias = sb[r2]; slot = 1; }
    __builtin_prefetch(w + DIM + lane * 16, 0, 0);   // next row -> global_prefetch_b8
    float acc = 0.f;
    #pragma unroll
    for (int j = 0; j < 4; ++j) {
      float4 wv = *(const float4*)(w + j * 128 + lane * 4);
      float4 hv = *(const float4*)(&sh[j * 128 + lane * 4]);
      acc += wv.x * hv.x + wv.y * hv.y + wv.z * hv.z + wv.w * hv.w;
    }
    #pragma unroll
    for (int off = 16; off > 0; off >>= 1) acc += __shfl_xor(acc, off, 32);
    if (lane == 0) {
      float zv = acc + bias;
      z[row] = zv;
      atomicMax(&smax[slot], enc_f32(zv));
    }
  }
  __syncthreads();
  if (tid < 2) atomicMax(&red[tid], smax[tid]);
}

// ---------------------------------------------------------------- K5: sum of exp
__global__ void k5_sumexp(const float* __restrict__ z, unsigned* __restrict__ red) {
  __shared__ float ssum[2];
  int tid = threadIdx.x;
  if (tid < 2) ssum[tid] = 0.f;
  __syncthreads();
  float mg = dec_f32(red[0]);
  float ms = dec_f32(red[1]);
  int i = blockIdx.x * 256 + tid;
  float eg = 0.f, es = 0.f;
  if (i < NTOT) {
    float zv = z[i];
    if (i < NGLOB) eg = __expf(zv - mg);
    else           es = __expf(zv - ms);
  }
  #pragma unroll
  for (int off = 16; off > 0; off >>= 1) {
    eg += __shfl_xor(eg, off, 32);
    es += __shfl_xor(es, off, 32);
  }
  if ((tid & 31) == 0) {
    if (eg != 0.f) atomicAdd(&ssum[0], eg);
    if (es != 0.f) atomicAdd(&ssum[1], es);
  }
  __syncthreads();
  float* redf = (float*)red;
  if (tid == 0 && ssum[0] != 0.f) atomicAdd(&redf[2], ssum[0]);
  if (tid == 1 && ssum[1] != 0.f) atomicAdd(&redf[3], ssum[1]);
}

// ---------------------------------------------------------------- K6: log-softmax out
__global__ void k6_finalize(const float* __restrict__ z, const unsigned* __restrict__ red,
                            float* __restrict__ out, int t) {
  int i = blockIdx.x * 256 + threadIdx.x;
  if (i >= NTOT) return;
  const float* redf = (const float*)red;
  float zv = z[i];
  if (i < NGLOB) {
    out[(size_t)t * NGLOB + i] = zv - dec_f32(red[0]) - __logf(redf[2]);
  } else {
    out[(size_t)T_STEPS * NGLOB + (size_t)t * NSENSE + (i - NGLOB)] =
        zv - dec_f32(red[1]) - __logf(redf[3]);
  }
}

// ----------------------------------------------------------------------------
extern "C" void kernel_launch(void* const* d_in, const int* in_sizes, int n_in,
                              void* d_out, int out_size, void* d_ws, size_t ws_size,
                              hipStream_t stream) {
  const int*   xidx  = (const int*)  d_in[0];
  const int*   eidx  = (const int*)  d_in[1];
  const float* X     = (const float*)d_in[2];
  const float* convW = (const float*)d_in[3];
  const float* W0    = (const float*)d_in[4];
  const float* gateW = (const float*)d_in[5];
  const float* gateU = (const float*)d_in[6];
  const float* gw    = (const float*)d_in[7];
  const float* gb    = (const float*)d_in[8];
  const float* sw    = (const float*)d_in[9];
  const float* sb    = (const float*)d_in[10];
  float* out = (float*)d_out;
  (void)in_sizes; (void)n_in; (void)out_size; (void)ws_size;

  char* ws = (char*)d_ws;
  size_t off = 0;
  auto wsalloc = [&](size_t bytes) -> void* {
    void* p = ws + off;
    off = (off + bytes + 255) & ~(size_t)255;
    return p;
  };
  float*    mem   = (float*)   wsalloc((size_t)ND * 4);              // GRU memory
  float*    gx    = (float*)   wsalloc((size_t)ND * 4);              // gathered feats f32
  _Float16* gxh   = (_Float16*)wsalloc((size_t)ND * 2);              // gathered feats f16
  _Float16* Wh    = (_Float16*)wsalloc((size_t)6 * DIM * DIM * 2);   // f16 weights
  float*    xw    = (float*)   wsalloc((size_t)6 * ND * 4);          // GEMM outputs
  float*    gatep = (float*)   wsalloc((size_t)64 * DIM * 4);        // gate partials
  float*    hbuf  = (float*)   wsalloc((size_t)DIM * 4);             // relu(rgcn[0])
  float*    z     = (float*)   wsalloc((size_t)NTOT * 4);            // logits scratch
  unsigned* red   = (unsigned*)wsalloc(64);                          // max/sum slots

  k0_init<<<(6 * DIM * DIM + 255) / 256, 256, 0, stream>>>(convW, W0, Wh, mem);
  for (int t = 0; t < T_STEPS; ++t) {
    k1_gather  <<<ND / 256, 256, 0, stream>>>(xidx, X, gx, gxh, t);
    k2_gemm    <<<48, 256, 0, stream>>>(gxh, Wh, xw);
    k25_gatep  <<<64, 256, 0, stream>>>(gx, gateW, mem, gateU, gatep);
    k3_combine <<<1, 512, 0, stream>>>(eidx, xw, gatep, mem, hbuf, red, t);
    k4_logits  <<<(NTOT + 63) / 64, 256, 0, stream>>>(hbuf, gw, gb, sw, sb, z, red);
    k5_sumexp  <<<(NTOT + 255) / 256, 256, 0, stream>>>(z, red);
    k6_finalize<<<(NTOT + 255) / 256, 256, 0, stream>>>(z, red, out, t);
  }
}
